// get_rec_selfsup_loss_21406117004172
// MI455X (gfx1250) — matrix operand order repaired and checked by reading, not compiled, not run on hardware
//
#include <hip/hip_runtime.h>
#include <hip/hip_bf16.h>

// Problem constants (match reference)
#define B_     16
#define C_     128
#define N_     2048
#define M_     2048
#define KCLS   16
#define MARGIN 0.5f
#define LCONT  1.0f
#define LREC   1.0f

typedef __attribute__((ext_vector_type(16))) __bf16 v16bf;
typedef __attribute__((ext_vector_type(8)))  float  v8f;

struct alignas(16) U128 { unsigned u[4]; };
struct B16F { U128 lo, hi; };   // 32 bytes -> v16bf

static __device__ __forceinline__ unsigned short f2bf(float f) {
    unsigned u = __float_as_uint(f);
    unsigned r = (u + 0x7FFFu + ((u >> 16) & 1u)) >> 16;   // RNE
    return (unsigned short)r;
}

static __device__ __forceinline__ unsigned rotl32(unsigned x, int r) {
    return (x << r) | (x >> (32 - r));
}

// Exact JAX Threefry-2x32 for key(42). One block cipher evaluation yields the
// uniforms for flat indices `flat` (< H) AND `flat + H` simultaneously.
// With flat = (b*N + n)*N + m and H = B*N*N/2, the pair is (b,n,m) / (b+8,n,m).
static __device__ __forceinline__ void tf_uniform_pair(unsigned flat, float& u_lo, float& u_hi) {
    const unsigned H = ((unsigned)B_ * N_ * N_) / 2u;   // 2^25
    unsigned x0 = flat, x1 = flat + H;
    const unsigned ks0 = 0u, ks1 = 42u, ks2 = 0u ^ 42u ^ 0x1BD11BDAu;
    x0 += ks0; x1 += ks1;
#define TFR(r) { x0 += x1; x1 = rotl32(x1, r); x1 ^= x0; }
    TFR(13) TFR(15) TFR(26) TFR(6)   x0 += ks1; x1 += ks2 + 1u;
    TFR(17) TFR(29) TFR(16) TFR(24)  x0 += ks2; x1 += ks0 + 2u;
    TFR(13) TFR(15) TFR(26) TFR(6)   x0 += ks0; x1 += ks1 + 3u;
    TFR(17) TFR(29) TFR(16) TFR(24)  x0 += ks1; x1 += ks2 + 4u;
    TFR(13) TFR(15) TFR(26) TFR(6)   x0 += ks2; x1 += ks0 + 5u;
#undef TFR
    u_lo = __uint_as_float((x0 >> 9) | 0x3F800000u) - 1.0f;
    u_hi = __uint_as_float((x1 >> 9) | 0x3F800000u) - 1.0f;
}

// ---------------- K0: zero accumulators ----------------
__global__ void k_init(float* accf, unsigned* posc) {
    if (threadIdx.x == 0) { accf[0] = 0.f; accf[1] = 0.f; accf[2] = 0.f; posc[0] = 0u; }
}

// ---------------- K1: normalize + transpose to bf16 fnT[b][n][c] ----------------
__global__ __launch_bounds__(256) void k_normalize(const float* __restrict__ feat,
                                                   unsigned short* __restrict__ fnT) {
    const int b = blockIdx.y;
    const int n = blockIdx.x * blockDim.x + threadIdx.x;
    const float* col = feat + (size_t)b * C_ * N_ + n;
    float s = 0.f;
#pragma unroll 4
    for (int c = 0; c < C_; ++c) { float v = col[(size_t)c * N_]; s += v * v; }
    const float inv = 1.0f / fmaxf(sqrtf(s), 1e-12f);
    unsigned short* out = fnT + ((size_t)b * N_ + n) * C_;
#pragma unroll 4
    for (int c = 0; c < C_; ++c) out[c] = f2bf(col[(size_t)c * N_] * inv);
}

// ---------------- K2: pos_count = sum_b sum_k count_k^2 ----------------
__global__ __launch_bounds__(256) void k_poscount(const int* __restrict__ target,
                                                  unsigned* __restrict__ posc) {
    __shared__ unsigned hist[KCLS];
    const int b = blockIdx.x;
    if (threadIdx.x < KCLS) hist[threadIdx.x] = 0u;
    __syncthreads();
    for (int n = threadIdx.x; n < N_; n += blockDim.x)
        atomicAdd(&hist[target[b * N_ + n] & (KCLS - 1)], 1u);
    __syncthreads();
    if (threadIdx.x < KCLS) { unsigned c = hist[threadIdx.x]; atomicAdd(posc, c * c); }
}

// GEMM strip: 16x64 output tile (4 WMMA tiles), K=128 in 4 steps of 32.
static __device__ __forceinline__ void gemm_strip(const unsigned short* __restrict__ base,
                                                  int n0, int m0, int row, int kh,
                                                  v8f acc[4]) {
    const unsigned short* arow = base + (size_t)(n0 + row) * C_;
#pragma unroll
    for (int ks = 0; ks < 4; ++ks) {
        const int c0 = ks * 32;
        // A fragment: 16-bit A 16x32 layout — lane(row,kh): K = j+8*kh (j<8), K = j+8+8*kh (j>=8)
        B16F abits;
        abits.lo = *(const U128*)(arow + c0 + kh * 8);
        abits.hi = *(const U128*)(arow + c0 + 16 + kh * 8);
        const v16bf a = __builtin_bit_cast(v16bf, abits);
#pragma unroll
        for (int t = 0; t < 4; ++t) {
            // B fragment: 32x16 — lane(col=row, khalf=kh): K = j+8*kh (j<8), K = j+8+8*kh (j>=8)
            const unsigned short* brow = base + (size_t)(m0 + t * 16 + row) * C_ + c0 + kh * 8;
            B16F bbits;
            bbits.lo = *(const U128*)(brow);
            bbits.hi = *(const U128*)(brow + 16);
            const v16bf bt = __builtin_bit_cast(v16bf, bbits);
            acc[t] = __builtin_amdgcn_wmma_f32_16x16x32_bf16(
                false, a, false, bt, (short)0, acc[t], false, false);
        }
    }
}

// ---------------- K3: fused pair_sim GEMM (bf16 WMMA) + contrastive loss ----------------
// One wave handles the SAME 16x64 strip for batches b and b+8, so every
// Threefry-2x32 evaluation supplies the sampling uniforms for both batches.
__global__ __launch_bounds__(256) void k_pairloss(const unsigned short* __restrict__ fnT,
                                                  const int* __restrict__ target,
                                                  const unsigned* __restrict__ posc,
                                                  float* __restrict__ cos_sum) {
    const int lane = threadIdx.x & 31;
    const int gw   = blockIdx.x * 8 + (threadIdx.x >> 5);
    const int WPB  = (N_ / 16) * (N_ / 64);      // strips per batch-pair = 4096
    const int b0   = gw / WPB;                   // 0..7
    const int b1   = b0 + B_ / 2;                // 8..15
    const int rem  = gw % WPB;
    const int n0   = (rem / (N_ / 64)) * 16;
    const int m0   = (rem % (N_ / 64)) * 64;
    const int row  = lane & 15;
    const int kh   = lane >> 4;

    v8f acc0[4] = {}, acc1[4] = {};
    gemm_strip(fnT + (size_t)b0 * N_ * C_, n0, m0, row, kh, acc0);
    gemm_strip(fnT + (size_t)b1 * N_ * C_, n0, m0, row, kh, acc1);

    // Fused epilogue: pair_target, diag mask, threefry sampling, margin loss
    const float posfrac = (float)(*posc) * (1.0f / ((float)B_ * (float)N_ * (float)N_));
    const float thr = 1.0f - posfrac;

    int tn0[8], tn1[8];
#pragma unroll
    for (int r = 0; r < 8; ++r) {
        const int n = n0 + r + kh * 8;
        tn0[r] = target[b0 * N_ + n];
        tn1[r] = target[b1 * N_ + n];
    }

    float sum = 0.f;
#pragma unroll
    for (int t = 0; t < 4; ++t) {
        const int m   = m0 + t * 16 + row;
        const int tm0 = target[b0 * N_ + m];
        const int tm1 = target[b1 * N_ + m];
#pragma unroll
        for (int r = 0; r < 8; ++r) {
            const int n = n0 + r + kh * 8;             // C/D layout: VGPR r -> M = r + 8*kh
            const unsigned flat = (unsigned)(b0 * N_ + n) * (unsigned)N_ + (unsigned)m; // < H
            float u0, u1;
            tf_uniform_pair(flat, u0, u1);
            // batch b0
            {
                const float sim = acc0[t][r];
                const bool  pos = (tn0[r] == tm0);
                const float l   = pos ? (1.0f - sim) : fmaxf(sim - MARGIN, 0.0f);
                if (n != m && (pos || u0 > thr)) sum += l;
            }
            // batch b0+8
            {
                const float sim = acc1[t][r];
                const bool  pos = (tn1[r] == tm1);
                const float l   = pos ? (1.0f - sim) : fmaxf(sim - MARGIN, 0.0f);
                if (n != m && (pos || u1 > thr)) sum += l;
            }
        }
    }
    // wave32 reduction, one atomic per wave
#pragma unroll
    for (int off = 16; off > 0; off >>= 1) sum += __shfl_xor(sum, off, 32);
    if (lane == 0) atomicAdd(cos_sum, sum);
}

// ---------------- K4a: per (b,m) min over n ----------------
__global__ __launch_bounds__(256) void k_chamfer_min_n(const float* __restrict__ pts,
                                                       const float* __restrict__ gt,
                                                       float* __restrict__ sumA) {
    __shared__ float sx[256], sy[256], sz[256];
    __shared__ float red[256];
    const int b = blockIdx.x / (M_ / 256);
    const int m = (blockIdx.x % (M_ / 256)) * 256 + threadIdx.x;
    const float* gb = gt + (size_t)b * 3 * M_;
    const float yx = gb[m], yy = gb[M_ + m], yz = gb[2 * M_ + m];
    const float* pb = pts + (size_t)b * N_ * 3;
    float best = 3.4e38f;
    for (int nb = 0; nb < N_; nb += 256) {
        const int j = nb + threadIdx.x;
        sx[threadIdx.x] = pb[j * 3 + 0];
        sy[threadIdx.x] = pb[j * 3 + 1];
        sz[threadIdx.x] = pb[j * 3 + 2];
        __syncthreads();
#pragma unroll 4
        for (int k = 0; k < 256; ++k) {
            const float dx = yx - sx[k], dy = yy - sy[k], dz = yz - sz[k];
            best = fminf(best, dx * dx + dy * dy + dz * dz);
        }
        __syncthreads();
    }
    red[threadIdx.x] = best; __syncthreads();
    for (int s = 128; s > 0; s >>= 1) {
        if (threadIdx.x < s) red[threadIdx.x] += red[threadIdx.x + s];
        __syncthreads();
    }
    if (threadIdx.x == 0) atomicAdd(sumA, red[0]);
}

// ---------------- K4b: per (b,n) min over m ----------------
__global__ __launch_bounds__(256) void k_chamfer_min_m(const float* __restrict__ pts,
                                                       const float* __restrict__ gt,
                                                       float* __restrict__ sumB) {
    __shared__ float sx[256], sy[256], sz[256];
    __shared__ float red[256];
    const int b = blockIdx.x / (N_ / 256);
    const int n = (blockIdx.x % (N_ / 256)) * 256 + threadIdx.x;
    const float* pb = pts + (size_t)b * N_ * 3 + (size_t)n * 3;
    const float xx = pb[0], xy = pb[1], xz = pb[2];
    const float* gb = gt + (size_t)b * 3 * M_;
    float best = 3.4e38f;
    for (int mb = 0; mb < M_; mb += 256) {
        const int j = mb + threadIdx.x;
        sx[threadIdx.x] = gb[j];
        sy[threadIdx.x] = gb[M_ + j];
        sz[threadIdx.x] = gb[2 * M_ + j];
        __syncthreads();
#pragma unroll 4
        for (int k = 0; k < 256; ++k) {
            const float dx = xx - sx[k], dy = xy - sy[k], dz = xz - sz[k];
            best = fminf(best, dx * dx + dy * dy + dz * dz);
        }
        __syncthreads();
    }
    red[threadIdx.x] = best; __syncthreads();
    for (int s = 128; s > 0; s >>= 1) {
        if (threadIdx.x < s) red[threadIdx.x] += red[threadIdx.x + s];
        __syncthreads();
    }
    if (threadIdx.x == 0) atomicAdd(sumB, red[0]);
}

// ---------------- K5: combine ----------------
__global__ void k_final(const float* __restrict__ accf, float* __restrict__ out) {
    if (threadIdx.x == 0) {
        const float cos_mean = accf[0] / ((float)B_ * (float)N_ * (float)N_);
        const float rec = accf[1] / ((float)B_ * (float)M_) + accf[2] / ((float)B_ * (float)N_);
        out[0] = LCONT * 0.5f * cos_mean + LREC * rec;
    }
}

extern "C" void kernel_launch(void* const* d_in, const int* in_sizes, int n_in,
                              void* d_out, int out_size, void* d_ws, size_t ws_size,
                              hipStream_t stream) {
    const float* feat   = (const float*)d_in[0];   // [B,C,N]
    const float* pts    = (const float*)d_in[1];   // [B,N,3]
    const float* gtpts  = (const float*)d_in[2];   // [B,3,M]
    const int*   target = (const int*)d_in[3];     // [B,N]
    float* out = (float*)d_out;

    // Workspace layout
    const size_t FNT_BYTES = (size_t)B_ * N_ * C_ * 2;          // 8 MB bf16 fnT
    unsigned short* fnT = (unsigned short*)d_ws;
    float*    accf = (float*)((char*)d_ws + FNT_BYTES);         // [0]=cos_sum [1]=sumA [2]=sumB
    unsigned* posc = (unsigned*)(accf + 3);

    k_init<<<1, 32, 0, stream>>>(accf, posc);
    k_normalize<<<dim3(N_ / 256, B_), 256, 0, stream>>>(feat, fnT);
    k_poscount<<<B_, 256, 0, stream>>>(target, posc);

    // (B/2)*(N/16)*(N/64) waves (each wave covers a batch pair), 8 waves/block
    const int nwaves = (B_ / 2) * (N_ / 16) * (N_ / 64);        // 32768
    k_pairloss<<<nwaves / 8, 256, 0, stream>>>(fnT, target, posc, accf + 0);

    k_chamfer_min_n<<<B_ * (M_ / 256), 256, 0, stream>>>(pts, gtpts, accf + 1);
    k_chamfer_min_m<<<B_ * (N_ / 256), 256, 0, stream>>>(pts, gtpts, accf + 2);

    k_final<<<1, 32, 0, stream>>>(accf, out);
}
